// Decoder_4578435138215
// MI455X (gfx1250) — matrix operand order repaired and checked by reading, not compiled, not run on hardware
//
#include <hip/hip_runtime.h>

typedef __bf16 bf16_t;
typedef __attribute__((ext_vector_type(16))) __bf16 v16bf;
typedef __attribute__((ext_vector_type(8)))  __bf16 v8bf;
typedef __attribute__((ext_vector_type(8)))  float  v8f;

#define BATCH 512

// ---------------------------------------------------------------- WMMA core
__device__ __forceinline__ v8f wmma_bf16(v16bf a, v16bf b, v8f c) {
  // D = A(16x32) * B(32x16) + C, fp32 accumulate
  return __builtin_amdgcn_wmma_f32_16x16x32_bf16(false, a, false, b,
                                                 (short)0, c, false, false);
}

// A fragment: 16x32 bf16 row-major, leading dim lda.
// lane m=L%16, half h=L/16. Layout: f[0..15] = A[m][h*16 .. h*16+15]
// -> two contiguous b128 loads.
__device__ __forceinline__ v16bf frag_a_16x32(const bf16_t* __restrict__ A,
                                              int lda, int lane) {
  int m = lane & 15, h = (lane >> 4) & 1;
  const bf16_t* p = A + m * lda + h * 16;
  v8bf lo = *(const v8bf*)(p);
  v8bf hi = *(const v8bf*)(p + 8);
  v16bf f;
#pragma unroll
  for (int e = 0; e < 8; ++e) { f[e] = lo[e]; f[8 + e] = hi[e]; }
  return f;
}

// B fragment from PRE-TRANSPOSED weights WT[N][K] (leading dim ldk):
// lane n=L%16, half h=L/16 needs K = h*16 .. h*16+15 of column n ->
// one contiguous 32-byte load (2x b128).
__device__ __forceinline__ v16bf frag_b_t(const bf16_t* __restrict__ WT,
                                          int ldk, int lane) {
  int n = lane & 15, h = (lane >> 4) & 1;
  return *(const v16bf*)(WT + n * ldk + h * 16);
}

// C/D: 8 VGPRs; lane n = L%16, M = (L/16)*8 + v
__device__ __forceinline__ void store_c_16x16(float* __restrict__ C, int ldc,
                                              int lane, v8f acc) {
  int n = lane & 15, h = (lane >> 4) & 1;
#pragma unroll
  for (int v = 0; v < 8; ++v) C[(h * 8 + v) * ldc + n] = acc[v];
}

// Paired fragments for the K=16 conv stage: two neighbor taps per K=32 WMMA.
__device__ __forceinline__ v16bf frag_a_pair(const bf16_t* __restrict__ A,
                                             int lda, int off0, int off1,
                                             int lane) {
  int m = lane & 15, h = (lane >> 4) & 1;
  const bf16_t* row = A + m * lda + h * 8;
  v8bf lo = *(const v8bf*)(row + off0);
  v8bf hi = {};
  if (off1 >= 0) hi = *(const v8bf*)(row + off1);
  v16bf f;
#pragma unroll
  for (int e = 0; e < 8; ++e) { f[e] = lo[e]; f[8 + e] = hi[e]; }
  return f;
}

// WT3 layout [27][N=16][K=16]; f[0..7] = tap0 K h*8..h*8+7, f[8..15] = tap1 K h*8..h*8+7.
__device__ __forceinline__ v16bf frag_b_pair_t(const bf16_t* __restrict__ WT,
                                               int tap0, int tap1, int lane) {
  int n = lane & 15, h = (lane >> 4) & 1;
  v8bf lo = {}, hi = {};
  if (tap0 >= 0) lo = *(const v8bf*)(WT + tap0 * 256 + n * 16 + h * 8);
  if (tap1 >= 0) hi = *(const v8bf*)(WT + tap1 * 256 + n * 16 + h * 8);
  v16bf f;
#pragma unroll
  for (int e = 0; e < 8; ++e) { f[e] = lo[e]; f[8 + e] = hi[e]; }
  return f;
}

// ---------------------------------------------------------------- GEMM stages

// tconv1: z[B,512] x Wt1^T[o][512N,512K] -> Y[B,8,512]
// grid(32,2,8) block 256; dual N-tile per wave (A-fragment reuse)
__global__ void k_tconv1(const bf16_t* __restrict__ A0,
                         const bf16_t* __restrict__ W, float* __restrict__ Y) {
  int lane = threadIdx.x & 31, wv = threadIdx.x >> 5;
  int rowB = blockIdx.x << 4;
  int col  = (blockIdx.y << 8) + (wv << 5);
  int o    = blockIdx.z;
  const bf16_t* Wo = W + (size_t)o * 512 * 512;
  v8f acc0 = {}, acc1 = {};
#pragma unroll 4
  for (int kk = 0; kk < 512; kk += 32) {
    v16bf a  = frag_a_16x32(A0 + (size_t)rowB * 512 + kk, 512, lane);
    v16bf b0 = frag_b_t(Wo + (size_t)col * 512 + kk, 512, lane);
    v16bf b1 = frag_b_t(Wo + (size_t)(col + 16) * 512 + kk, 512, lane);
    acc0 = wmma_bf16(a, b0, acc0);
    acc1 = wmma_bf16(a, b1, acc1);
  }
  float* Cp = Y + (size_t)rowB * 4096 + (size_t)o * 512 + col;
  store_c_16x16(Cp, 4096, lane, acc0);
  store_c_16x16(Cp + 16, 4096, lane, acc1);
}

// conv3 on 2^3, 512ch: all 8 inputs feed every output.
// grid(32,2,8) block 256; dual N-tile per wave
__global__ void k_conv1(const bf16_t* __restrict__ A1,
                        const bf16_t* __restrict__ WC, float* __restrict__ Y) {
  int lane = threadIdx.x & 31, wv = threadIdx.x >> 5;
  int rowB = blockIdx.x << 4;
  int col  = (blockIdx.y << 8) + (wv << 5);
  int p = blockIdx.z;
  int pd = p >> 2, ph = (p >> 1) & 1, pw = p & 1;
  v8f acc0 = {}, acc1 = {};
  for (int q = 0; q < 8; ++q) {
    int qd = q >> 2, qh = (q >> 1) & 1, qw = q & 1;
    int tap = ((qd - pd + 1) * 3 + (qh - ph + 1)) * 3 + (qw - pw + 1);
    const bf16_t* Wt = WC + (size_t)tap * 512 * 512;
    const bf16_t* Aq = A1 + (size_t)rowB * 4096 + (size_t)q * 512;
#pragma unroll 4
    for (int kk = 0; kk < 512; kk += 32) {
      __builtin_prefetch(Aq + kk + 128, 0, 1);
      v16bf a  = frag_a_16x32(Aq + kk, 4096, lane);
      v16bf b0 = frag_b_t(Wt + (size_t)col * 512 + kk, 512, lane);
      v16bf b1 = frag_b_t(Wt + (size_t)(col + 16) * 512 + kk, 512, lane);
      acc0 = wmma_bf16(a, b0, acc0);
      acc1 = wmma_bf16(a, b1, acc1);
    }
  }
  float* Cp = Y + (size_t)rowB * 4096 + (size_t)p * 512 + col;
  store_c_16x16(Cp, 4096, lane, acc0);
  store_c_16x16(Cp + 16, 4096, lane, acc1);
}

// tconv2: A[B,8,512] x Wt2^T[o][64N,512K] -> Y[B,64,64]. grid(32,64) block 128
__global__ void k_tconv2(const bf16_t* __restrict__ A2,
                         const bf16_t* __restrict__ W, float* __restrict__ Y) {
  int lane = threadIdx.x & 31, wv = threadIdx.x >> 5;
  int rowB = blockIdx.x << 4;
  int col  = wv << 4;
  int v = blockIdx.y;
  int d = v >> 4, h = (v >> 2) & 3, w = v & 3;
  int q = ((d >> 1) << 2) + ((h >> 1) << 1) + (w >> 1);
  int o = ((d & 1) << 2) + ((h & 1) << 1) + (w & 1);
  const bf16_t* Wo = W + (size_t)o * 64 * 512;
  const bf16_t* Aq = A2 + (size_t)rowB * 4096 + (size_t)q * 512;
  v8f acc = {};
#pragma unroll 4
  for (int kk = 0; kk < 512; kk += 32) {
    v16bf a = frag_a_16x32(Aq + kk, 4096, lane);
    v16bf b = frag_b_t(Wo + (size_t)col * 512 + kk, 512, lane);
    acc = wmma_bf16(a, b, acc);
  }
  store_c_16x16(Y + (size_t)rowB * 4096 + (size_t)v * 64 + col, 4096, lane, acc);
}

// conv3 on 4^3, 64ch; WC2^T[tap][64N,64K]. grid(32,64) block 128
__global__ void k_conv2(const bf16_t* __restrict__ A3,
                        const bf16_t* __restrict__ WC, float* __restrict__ Y) {
  int lane = threadIdx.x & 31, wv = threadIdx.x >> 5;
  int rowB = blockIdx.x << 4;
  int col  = wv << 4;
  int p = blockIdx.y;
  int pd = p >> 4, ph = (p >> 2) & 3, pw = p & 3;
  v8f acc = {};
  for (int dz = -1; dz <= 1; ++dz)
    for (int dy = -1; dy <= 1; ++dy)
      for (int dx = -1; dx <= 1; ++dx) {
        int qd = pd + dz, qh = ph + dy, qw = pw + dx;
        if ((unsigned)qd > 3u || (unsigned)qh > 3u || (unsigned)qw > 3u) continue;
        int q = (qd << 4) + (qh << 2) + qw;
        int tap = ((dz + 1) * 3 + (dy + 1)) * 3 + (dx + 1);
        const bf16_t* Wt = WC + (size_t)tap * 64 * 64;
        const bf16_t* Aq = A3 + (size_t)rowB * 4096 + (size_t)q * 64;
#pragma unroll
        for (int kk = 0; kk < 64; kk += 32) {
          v16bf a = frag_a_16x32(Aq + kk, 4096, lane);
          v16bf b = frag_b_t(Wt + (size_t)col * 64 + kk, 64, lane);
          acc = wmma_bf16(a, b, acc);
        }
      }
  store_c_16x16(Y + (size_t)rowB * 4096 + (size_t)p * 64 + col, 4096, lane, acc);
}

// tconv3: A[B,64,64] x Wt3^T[o][16N,64K] -> Y[B,512,16]. grid(4,512) block 256
__global__ void k_tconv3(const bf16_t* __restrict__ A4,
                         const bf16_t* __restrict__ W, float* __restrict__ Y) {
  int lane = threadIdx.x & 31, wv = threadIdx.x >> 5;
  int rowB = ((blockIdx.x << 3) + wv) << 4;
  int v = blockIdx.y;
  int d = v >> 6, h = (v >> 3) & 7, w = v & 7;
  int q = ((d >> 1) << 4) + ((h >> 1) << 2) + (w >> 1);
  int o = ((d & 1) << 2) + ((h & 1) << 1) + (w & 1);
  const bf16_t* Wo = W + (size_t)o * 16 * 64;
  const bf16_t* Aq = A4 + (size_t)rowB * 4096 + (size_t)q * 64;
  v8f acc = {};
#pragma unroll
  for (int kk = 0; kk < 64; kk += 32) {
    v16bf a = frag_a_16x32(Aq + kk, 4096, lane);
    v16bf b = frag_b_t(Wo + kk, 64, lane);
    acc = wmma_bf16(a, b, acc);
  }
  store_c_16x16(Y + (size_t)rowB * 8192 + (size_t)v * 16, 8192, lane, acc);
}

// conv3 on 8^3, 16ch: two taps per K=32 WMMA; WC3^T[27][16N][16K].
// grid(4,512) block 256
__global__ void k_conv3(const bf16_t* __restrict__ A5,
                        const bf16_t* __restrict__ WC, float* __restrict__ Y) {
  int lane = threadIdx.x & 31, wv = threadIdx.x >> 5;
  int rowB = ((blockIdx.x << 3) + wv) << 4;
  int p = blockIdx.y;
  int pd = p >> 6, ph = (p >> 3) & 7, pw = p & 7;
  int qoff[27], qtap[27], nq = 0;
  for (int dz = -1; dz <= 1; ++dz)
    for (int dy = -1; dy <= 1; ++dy)
      for (int dx = -1; dx <= 1; ++dx) {
        int qd = pd + dz, qh = ph + dy, qw = pw + dx;
        if ((unsigned)qd > 7u || (unsigned)qh > 7u || (unsigned)qw > 7u) continue;
        qoff[nq] = ((qd << 6) + (qh << 3) + qw) * 16;
        qtap[nq] = ((dz + 1) * 3 + (dy + 1)) * 3 + (dx + 1);
        ++nq;
      }
  const bf16_t* Ab = A5 + (size_t)rowB * 8192;
  v8f acc = {};
  for (int i = 0; i < nq; i += 2) {
    int o1 = (i + 1 < nq) ? qoff[i + 1] : -1;
    int t1 = (i + 1 < nq) ? qtap[i + 1] : -1;
    v16bf a = frag_a_pair(Ab, 8192, qoff[i], o1, lane);
    v16bf b = frag_b_pair_t(WC, qtap[i], t1, lane);
    acc = wmma_bf16(a, b, acc);
  }
  store_c_16x16(Y + (size_t)rowB * 8192 + (size_t)p * 16, 8192, lane, acc);
}

// ---------------------------------------------------------------- elementwise

__global__ void k_f2bf(const float* __restrict__ in, bf16_t* __restrict__ out,
                       int n) {
  for (int i = blockIdx.x * blockDim.x + threadIdx.x; i < n;
       i += gridDim.x * blockDim.x)
    out[i] = (bf16_t)in[i];
}

// fp32 [T][K][N] -> bf16 transposed [T][N][K] (B-fragment friendly layout)
__global__ void k_f2bf_t(const float* __restrict__ in, bf16_t* __restrict__ out,
                         int T, int K, int N) {
  int total = T * K * N;
  for (int i = blockIdx.x * blockDim.x + threadIdx.x; i < total;
       i += gridDim.x * blockDim.x) {
    int t = i / (K * N);
    int rem = i - t * K * N;
    int k = rem / N;
    int n = rem - k * N;
    out[(size_t)t * N * K + (size_t)n * K + k] = (bf16_t)in[i];
  }
}

// per-channel masked sums: one block per channel, no atomics.
__global__ void k_bnstats(const float* __restrict__ Y,
                          const float* __restrict__ mask, int rows, int C,
                          float* __restrict__ sums, float* __restrict__ sumsq,
                          float* __restrict__ cnt) {
  int c = blockIdx.x;
  __shared__ float s1[256], s2[256], sc[256];
  float a1 = 0.f, a2 = 0.f, ac = 0.f;
  for (int r = threadIdx.x; r < rows; r += blockDim.x) {
    float m = mask ? mask[r] : 1.0f;
    float x = Y[(size_t)r * C + c];
    a1 += x * m;
    a2 += x * x * m;
    ac += m;
  }
  s1[threadIdx.x] = a1; s2[threadIdx.x] = a2; sc[threadIdx.x] = ac;
  __syncthreads();
  for (int s = 128; s > 0; s >>= 1) {
    if (threadIdx.x < s) {
      s1[threadIdx.x] += s1[threadIdx.x + s];
      s2[threadIdx.x] += s2[threadIdx.x + s];
      sc[threadIdx.x] += sc[threadIdx.x + s];
    }
    __syncthreads();
  }
  if (threadIdx.x == 0) {
    sums[c] = s1[0];
    sumsq[c] = s2[0];
    if (c == 0) cnt[0] = sc[0];
  }
}

// BN + ELU + mask, emit bf16 for the next WMMA stage.
__global__ void k_bnapply(const float* __restrict__ Y,
                          const float* __restrict__ mask,
                          const float* __restrict__ g,
                          const float* __restrict__ be,
                          const float* __restrict__ sums,
                          const float* __restrict__ sumsq,
                          const float* __restrict__ cnt,
                          bf16_t* __restrict__ Aout, int rows, int C) {
  float n = fmaxf(cnt[0], 1.0f);
  size_t total = (size_t)rows * (size_t)C;
  size_t stride = (size_t)gridDim.x * blockDim.x;
  for (size_t i = blockIdx.x * (size_t)blockDim.x + threadIdx.x; i < total;
       i += stride) {
    int c = (int)(i % C);
    int r = (int)(i / C);
    float m = mask ? mask[r] : 1.0f;
    float mu = sums[c] / n;
    float var = sumsq[c] / n - mu * mu;
    float y = (Y[i] - mu) * rsqrtf(var + 1e-5f) * g[c] + be[c];
    y *= m;
    y = (y > 0.f) ? y : (__expf(y) - 1.f);
    y *= m;
    Aout[i] = (bf16_t)y;
  }
}

// cls1 + occ32 target + keep1; zero pruned rows of A4 in place.
__global__ void k_cls1(bf16_t* __restrict__ A4, const float* __restrict__ wcls,
                       const float* __restrict__ bcls,
                       const int* __restrict__ occ, float* __restrict__ out_cls1,
                       float* __restrict__ out_tgt1, float* __restrict__ keep1) {
  int i = blockIdx.x * blockDim.x + threadIdx.x;  // b*64 + v4
  if (i >= BATCH * 64) return;
  int b = i >> 6, v = i & 63;
  bf16_t* row = A4 + (size_t)i * 64;
  float acc = 0.f;
  for (int c = 0; c < 64; ++c) acc += (float)row[c] * wcls[c];
  float cls = acc + bcls[0];
  int d = v >> 4, h = (v >> 2) & 3, w = v & 3;
  int t = 0;
  for (int dz = 0; dz < 2; ++dz)
    for (int dy = 0; dy < 2; ++dy)
      for (int dx = 0; dx < 2; ++dx)
        t |= (occ[((b * 8 + 2 * d + dz) * 8 + 2 * h + dy) * 8 + 2 * w + dx] > 0);
  float keep = ((cls > 0.f) || t) ? 1.f : 0.f;
  out_cls1[i] = cls;
  out_tgt1[i] = (float)t;
  keep1[i] = keep;
  if (keep == 0.f)
    for (int c = 0; c < 64; ++c) row[c] = (bf16_t)0.f;
}

// m2 = up(keep1): child mask on 8^3 from parent 4^3
__global__ void k_upmask(const float* __restrict__ keep1,
                         float* __restrict__ mask2) {
  int i = blockIdx.x * blockDim.x + threadIdx.x;  // b*512 + v8
  if (i >= BATCH * 512) return;
  int b = i >> 9, v = i & 511;
  int d = v >> 6, h = (v >> 3) & 7, w = v & 7;
  int parent = ((d >> 1) << 4) + ((h >> 1) << 2) + (w >> 1);
  mask2[i] = keep1[b * 64 + parent];
}

// final BN+ELU+mask, cls2, tgt2, keep2, out2
__global__ void k_final(const float* __restrict__ Y6,
                        const float* __restrict__ mask2,
                        const float* __restrict__ g,
                        const float* __restrict__ be,
                        const float* __restrict__ sums,
                        const float* __restrict__ sumsq,
                        const float* __restrict__ cnt,
                        const float* __restrict__ wcls,
                        const float* __restrict__ bcls,
                        const int* __restrict__ occ,
                        float* __restrict__ out_cls2,
                        float* __restrict__ out_tgt2, float* __restrict__ out2) {
  int i = blockIdx.x * blockDim.x + threadIdx.x;  // b*512 + v8
  if (i >= BATCH * 512) return;
  float m = mask2[i];
  float n = fmaxf(cnt[0], 1.f);
  const float* row = Y6 + (size_t)i * 16;
  float act[16];
  float acc = 0.f;
  for (int c = 0; c < 16; ++c) {
    float mu = sums[c] / n;
    float var = sumsq[c] / n - mu * mu;
    float y = (row[c] - mu) * rsqrtf(var + 1e-5f) * g[c] + be[c];
    y *= m;
    y = (y > 0.f) ? y : (__expf(y) - 1.f);
    y *= m;
    act[c] = y;
    acc += y * wcls[c];
  }
  float cls = (acc + bcls[0]) * m;
  int b = i >> 9, v = i & 511;
  int tg = (occ[b * 512 + v] > 0) && (m > 0.f);
  float keep = (((cls > 0.f) || tg) && (m > 0.f)) ? 1.f : 0.f;
  out_cls2[i] = cls;
  out_tgt2[i] = (float)tg;
  for (int c = 0; c < 16; ++c) out2[(size_t)i * 16 + c] = act[c] * keep;
}

// ---------------------------------------------------------------- launch
extern "C" void kernel_launch(void* const* d_in, const int* in_sizes, int n_in,
                              void* d_out, int out_size, void* d_ws,
                              size_t ws_size, hipStream_t stream) {
  (void)in_sizes; (void)n_in; (void)out_size; (void)ws_size;
  const float* z     = (const float*)d_in[0];
  const int*   occ   = (const int*)d_in[1];
  const float* w_t1  = (const float*)d_in[2];
  const float* g_t1  = (const float*)d_in[3];
  const float* be_t1 = (const float*)d_in[4];
  const float* w_c1  = (const float*)d_in[5];
  const float* g_c1  = (const float*)d_in[6];
  const float* be_c1 = (const float*)d_in[7];
  const float* w_t2  = (const float*)d_in[8];
  const float* g_t2  = (const float*)d_in[9];
  const float* be_t2 = (const float*)d_in[10];
  const float* w_c2  = (const float*)d_in[11];
  const float* g_c2  = (const float*)d_in[12];
  const float* be_c2 = (const float*)d_in[13];
  const float* w_cls1 = (const float*)d_in[14];
  const float* b_cls1 = (const float*)d_in[15];
  const float* w_t3  = (const float*)d_in[16];
  const float* g_t3  = (const float*)d_in[17];
  const float* be_t3 = (const float*)d_in[18];
  const float* w_c3  = (const float*)d_in[19];
  const float* g_c3  = (const float*)d_in[20];
  const float* be_c3 = (const float*)d_in[21];
  const float* w_cls2 = (const float*)d_in[22];
  const float* b_cls2 = (const float*)d_in[23];

  // output tuple (cls1, cls2, tgt1, tgt2, out2) concatenated, fp32
  float* out = (float*)d_out;
  float* o_cls1 = out;                  // 512*64
  float* o_cls2 = out + 32768;          // 512*512
  float* o_tgt1 = out + 294912;         // 512*64
  float* o_tgt2 = out + 327680;         // 512*512
  float* o_out2 = out + 589824;         // 512*512*16

  // workspace carve-up
  char* base = (char*)d_ws;
  size_t off = 0;
  auto alloc = [&](size_t bytes) -> void* {
    void* p = base + off;
    off = (off + bytes + 255) & ~(size_t)255;
    return p;
  };
  bf16_t* wt1 = (bf16_t*)alloc((size_t)8 * 512 * 512 * 2);   // [o][N][K]
  bf16_t* wc1 = (bf16_t*)alloc((size_t)27 * 512 * 512 * 2);  // [tap][N][K]
  bf16_t* wt2 = (bf16_t*)alloc((size_t)8 * 64 * 512 * 2);    // [o][N=64][K=512]
  bf16_t* wc2 = (bf16_t*)alloc((size_t)27 * 64 * 64 * 2);
  bf16_t* wt3 = (bf16_t*)alloc((size_t)8 * 16 * 64 * 2);     // [o][N=16][K=64]
  bf16_t* wc3 = (bf16_t*)alloc((size_t)27 * 16 * 16 * 2);
  bf16_t* a0  = (bf16_t*)alloc((size_t)BATCH * 512 * 2);
  bf16_t* Aa  = (bf16_t*)alloc((size_t)BATCH * 8192 * 2);    // 8 MB
  bf16_t* Ab  = (bf16_t*)alloc((size_t)BATCH * 8192 * 2);    // 8 MB
  float*  Ya  = (float*)alloc((size_t)BATCH * 8192 * 4);     // 16 MB
  float*  Yb  = (float*)alloc((size_t)BATCH * 8192 * 4);     // 16 MB
  float* sums  = (float*)alloc(512 * 4);
  float* sumsq = (float*)alloc(512 * 4);
  float* cnt   = (float*)alloc(256);
  float* keep1 = (float*)alloc((size_t)BATCH * 64 * 4);
  float* mask2 = (float*)alloc((size_t)BATCH * 512 * 4);

  // weight conversion: fp32 [T][K][N] -> bf16 transposed [T][N][K]
  k_f2bf_t<<<2048, 256, 0, stream>>>(w_t1, wt1, 8, 512, 512);
  k_f2bf_t<<<4096, 256, 0, stream>>>(w_c1, wc1, 27, 512, 512);
  k_f2bf_t<<<512, 256, 0, stream>>>(w_t2, wt2, 8, 512, 64);
  k_f2bf_t<<<256, 256, 0, stream>>>(w_c2, wc2, 27, 64, 64);
  k_f2bf_t<<<32, 256, 0, stream>>>(w_t3, wt3, 8, 64, 16);
  k_f2bf_t<<<27, 256, 0, stream>>>(w_c3, wc3, 27, 16, 16);
  k_f2bf<<<512, 256, 0, stream>>>(z, a0, BATCH * 512);

  // ---- block1: tconv(512->512), BN, ELU
  k_tconv1<<<dim3(32, 2, 8), 256, 0, stream>>>(a0, wt1, Ya);
  k_bnstats<<<512, 256, 0, stream>>>(Ya, nullptr, BATCH * 8, 512, sums, sumsq, cnt);
  k_bnapply<<<4096, 256, 0, stream>>>(Ya, nullptr, g_t1, be_t1, sums, sumsq, cnt,
                                      Aa, BATCH * 8, 512);
  // conv3(512->512), BN, ELU
  k_conv1<<<dim3(32, 2, 8), 256, 0, stream>>>(Aa, wc1, Yb);
  k_bnstats<<<512, 256, 0, stream>>>(Yb, nullptr, BATCH * 8, 512, sums, sumsq, cnt);
  k_bnapply<<<4096, 256, 0, stream>>>(Yb, nullptr, g_c1, be_c1, sums, sumsq, cnt,
                                      Ab, BATCH * 8, 512);
  // tconv(512->64), BN, ELU
  k_tconv2<<<dim3(32, 64), 128, 0, stream>>>(Ab, wt2, Ya);
  k_bnstats<<<64, 256, 0, stream>>>(Ya, nullptr, BATCH * 64, 64, sums, sumsq, cnt);
  k_bnapply<<<4096, 256, 0, stream>>>(Ya, nullptr, g_t2, be_t2, sums, sumsq, cnt,
                                      Aa, BATCH * 64, 64);
  // conv3(64->64), BN, ELU
  k_conv2<<<dim3(32, 64), 128, 0, stream>>>(Aa, wc2, Yb);
  k_bnstats<<<64, 256, 0, stream>>>(Yb, nullptr, BATCH * 64, 64, sums, sumsq, cnt);
  k_bnapply<<<4096, 256, 0, stream>>>(Yb, nullptr, g_c2, be_c2, sums, sumsq, cnt,
                                      Ab, BATCH * 64, 64);
  // cls1 / tgt1 / keep1, prune A4 rows
  k_cls1<<<128, 256, 0, stream>>>(Ab, w_cls1, b_cls1, occ, o_cls1, o_tgt1, keep1);
  k_upmask<<<1024, 256, 0, stream>>>(keep1, mask2);

  // ---- block2: tconv(64->16), masked BN, ELU
  k_tconv3<<<dim3(4, 512), 256, 0, stream>>>(Ab, wt3, Ya);
  k_bnstats<<<16, 256, 0, stream>>>(Ya, mask2, BATCH * 512, 16, sums, sumsq, cnt);
  k_bnapply<<<4096, 256, 0, stream>>>(Ya, mask2, g_t3, be_t3, sums, sumsq, cnt,
                                      Aa, BATCH * 512, 16);
  // conv3(16->16), masked BN, ELU, cls2/tgt2/out2
  k_conv3<<<dim3(4, 512), 256, 0, stream>>>(Aa, wc3, Yb);
  k_bnstats<<<16, 256, 0, stream>>>(Yb, mask2, BATCH * 512, 16, sums, sumsq, cnt);
  k_final<<<1024, 256, 0, stream>>>(Yb, mask2, g_c3, be_c3, sums, sumsq, cnt,
                                    w_cls2, b_cls2, occ, o_cls2, o_tgt2, o_out2);
}
